// TextMambaFusion_57853209477736
// MI455X (gfx1250) — matrix operand order repaired and checked by reading, not compiled, not run on hardware
//
#include <hip/hip_runtime.h>
#include <math.h>

// ---------------- model constants ----------------
constexpr int Lc = 4, Dc = 512, DIc = 1024, DSc = 16, Vc = 50257;
constexpr int Bc = 2, Sc = 1024;
constexpr int Mrows = Bc * Sc;           // 2048 token rows
constexpr float EPSc = 1e-5f;

// ---------------- vector types for WMMA ----------------
typedef __attribute__((ext_vector_type(8)))  float  v8f;
typedef __attribute__((ext_vector_type(16))) __bf16 v16bf;
typedef __attribute__((ext_vector_type(8)))  __bf16 bf16x8;

// bf16 <-> f32 via explicit bit manipulation (round-to-nearest-even)
__device__ inline float bf2f(__bf16 b) {
    unsigned short s = __builtin_bit_cast(unsigned short, b);
    unsigned u = ((unsigned)s) << 16;
    return __builtin_bit_cast(float, u);
}
__device__ inline __bf16 f2bf(float f) {
    unsigned u = __builtin_bit_cast(unsigned, f);
    unsigned r = u + 0x7FFFu + ((u >> 16) & 1u);
    return __builtin_bit_cast(__bf16, (unsigned short)(r >> 16));
}

// =====================================================================
// GEMM: C[M,N] = A[M,K] @ B[K,N] (+bias[N]) (+Cres[M,N]), fp32 in/out.
// bf16x3 split (hi/lo) through LDS, v_wmma_f32_16x16x32_bf16, fp32 accum.
// Block = 256 threads = 8 waves (2x4), block tile 32(M) x 64(N), KC=32.
// =====================================================================
__global__ __launch_bounds__(256) void k_gemm_bf16x3(
    const float* __restrict__ A, int lda,
    const float* __restrict__ Bm, int ldb,
    float* __restrict__ C, int ldc,
    int Md, int Nd, int Kd,
    const float* __restrict__ bias, const float* __restrict__ Cres)
{
    constexpr int BM = 32, BN = 64, KC = 32, AP = 40, BP = 40; // pitch 40 elems = 80B (16B aligned rows, conflict-free)
    __shared__ alignas(16) __bf16 Ah[BM * AP];
    __shared__ alignas(16) __bf16 Al[BM * AP];
    __shared__ alignas(16) __bf16 Bh[BN * BP];   // stored transposed: [n][k]
    __shared__ alignas(16) __bf16 Bl[BN * BP];

    const int tid  = threadIdx.x;
    const int lane = tid & 31, wv = tid >> 5;
    const int wm = wv >> 2, wn = wv & 3;         // 2 x 4 wave grid
    const int hf = lane >> 4, q = lane & 15;     // half-wave, index within half
    const int m0 = blockIdx.y * BM, n0 = blockIdx.x * BN;

    v8f acc = {};

    for (int k0 = 0; k0 < Kd; k0 += KC) {
        __syncthreads();
        // ---- stage A tile (BM x KC), split into hi/lo bf16 ----
        #pragma unroll
        for (int it = 0; it < (BM * KC) / 256; ++it) {
            int i = tid + it * 256;
            int r = i >> 5, kk = i & 31;
            float v = A[(size_t)(m0 + r) * lda + (k0 + kk)];
            __bf16 hi = f2bf(v);
            __bf16 lo = f2bf(v - bf2f(hi));
            Ah[r * AP + kk] = hi;
            Al[r * AP + kk] = lo;
        }
        // ---- stage B tile (KC x BN), transposed to [n][k], split hi/lo ----
        #pragma unroll
        for (int it = 0; it < (BN * KC) / 256; ++it) {
            int i = tid + it * 256;
            int nn = i & 63, kk = i >> 6;        // consecutive tid -> consecutive n (coalesced)
            int col = n0 + nn;
            float v = (col < Nd) ? Bm[(size_t)(k0 + kk) * ldb + col] : 0.0f;
            __bf16 hi = f2bf(v);
            __bf16 lo = f2bf(v - bf2f(hi));
            Bh[nn * BP + kk] = hi;
            Bl[nn * BP + kk] = lo;
        }
        __syncthreads();

        // prefetch next K-chunk into cache (global_prefetch_b8)
        if (k0 + KC < Kd) {
            __builtin_prefetch(&A[(size_t)(m0 + (tid >> 3)) * lda + (k0 + KC + (tid & 7) * 4)], 0, 3);
            __builtin_prefetch(&Bm[(size_t)(k0 + KC + (tid >> 6)) * ldb + n0 + (tid & 63)], 0, 3);
        }

        // ---- build fragments per ISA VGPR layouts ----
        // A (16-bit 16x32): elems 0..7 -> K = 8*half+e ; elems 8..15 -> K = 16+8*half+e
        // B (16-bit 32x16): elems 0..7 -> K = 8*half+e ; elems 8..15 -> K = 16+8*half+e, N = lane&15 (LDS holds B^T so this is a row run)
        union Frag { bf16x8 h8[2]; v16bf v; };
        const __bf16* arh = &Ah[(wm * 16 + q) * AP];
        const __bf16* arl = &Al[(wm * 16 + q) * AP];
        const __bf16* brh = &Bh[(wn * 16 + q) * BP];
        const __bf16* brl = &Bl[(wn * 16 + q) * BP];

        Frag fah, fal, fbh, fbl;
        fah.h8[0] = *(const bf16x8*)(arh + 8 * hf);
        fah.h8[1] = *(const bf16x8*)(arh + 16 + 8 * hf);
        fal.h8[0] = *(const bf16x8*)(arl + 8 * hf);
        fal.h8[1] = *(const bf16x8*)(arl + 16 + 8 * hf);
        fbh.h8[0] = *(const bf16x8*)(brh + 8 * hf);
        fbh.h8[1] = *(const bf16x8*)(brh + 16 + 8 * hf);
        fbl.h8[0] = *(const bf16x8*)(brl + 8 * hf);
        fbl.h8[1] = *(const bf16x8*)(brl + 16 + 8 * hf);

        // bf16x3: hi*hi + hi*lo + lo*hi (lo*lo term ~2^-16 relative, dropped)
        acc = __builtin_amdgcn_wmma_f32_16x16x32_bf16(false, fah.v, false, fbh.v, (short)0, acc, false, false);
        acc = __builtin_amdgcn_wmma_f32_16x16x32_bf16(false, fah.v, false, fbl.v, (short)0, acc, false, false);
        acc = __builtin_amdgcn_wmma_f32_16x16x32_bf16(false, fal.v, false, fbh.v, (short)0, acc, false, false);
    }

    // ---- epilogue: C/D layout: VGPR e holds M = e + 8*half, N = lane&15 ----
    #pragma unroll
    for (int e = 0; e < 8; ++e) {
        int row = m0 + wm * 16 + e + 8 * hf;
        int col = n0 + wn * 16 + q;
        if (row < Md && col < Nd) {
            float o = acc[e];
            if (bias) o += bias[col];
            if (Cres) o += Cres[(size_t)row * ldc + col];
            C[(size_t)row * ldc + col] = o;
        }
    }
}

// =====================================================================
// Embedding + positional: h[b,s,:] = emb[x[b,s],:] + pos[0,s,:]
// =====================================================================
__global__ __launch_bounds__(256) void k_embed(
    const int* __restrict__ x, const float* __restrict__ emb,
    const float* __restrict__ pos, float* __restrict__ h)
{
    int idx = blockIdx.x * 256 + threadIdx.x;      // over Mrows*Dc
    int row = idx / Dc, d = idx % Dc;
    int s = row % Sc;
    int tok = x[row];
    h[idx] = emb[(size_t)tok * Dc + d] + pos[(size_t)s * Dc + d];
}

// =====================================================================
// Depthwise causal conv(w=4) + temb add + SiLU + LayerNorm + ||xc|| clamp.
// One block per (b,s) row; 256 threads x 4 channels each.
// =====================================================================
__global__ __launch_bounds__(256) void k_conv_ln(
    const float* __restrict__ proj, const int* __restrict__ t,
    const float* __restrict__ temb,
    const float* __restrict__ cw, const float* __restrict__ cb,
    const float* __restrict__ lg, const float* __restrict__ lb,
    float* __restrict__ xc, float* __restrict__ xnrm)
{
    __shared__ float r1[256], r2[256];
    const int row = blockIdx.x;                    // b*S + s
    const int b = row / Sc, s = row % Sc;
    const int tid = threadIdx.x;
    const int trow = t[b];

    float val[4];
    float lsum = 0.f, lsq = 0.f;
    #pragma unroll
    for (int j = 0; j < 4; ++j) {
        int c = tid + j * 256;
        float te = temb[(size_t)trow * DIc + c];
        float a = cb[c];
        #pragma unroll
        for (int jj = 0; jj < 4; ++jj) {
            int sp = s - 3 + jj;
            if (sp >= 0)
                a += cw[c * 4 + jj] * (proj[(size_t)(b * Sc + sp) * (2 * DIc) + c] + te);
        }
        float v = a / (1.0f + expf(-a));           // SiLU
        val[j] = v;
        lsum += v; lsq += v * v;
    }
    r1[tid] = lsum; r2[tid] = lsq;
    __syncthreads();
    for (int o = 128; o > 0; o >>= 1) {
        if (tid < o) { r1[tid] += r1[tid + o]; r2[tid] += r2[tid + o]; }
        __syncthreads();
    }
    float mu = r1[0] * (1.0f / DIc);
    float var = r2[0] * (1.0f / DIc) - mu * mu;
    float rs = rsqrtf(var + EPSc);

    float lsq2 = 0.f;
    #pragma unroll
    for (int j = 0; j < 4; ++j) {
        int c = tid + j * 256;
        float v = (val[j] - mu) * rs * lg[c] + lb[c];
        xc[(size_t)row * DIc + c] = v;
        lsq2 += v * v;
    }
    __syncthreads();
    r1[tid] = lsq2;
    __syncthreads();
    for (int o = 128; o > 0; o >>= 1) {
        if (tid < o) r1[tid] += r1[tid + o];
        __syncthreads();
    }
    if (tid == 0) xnrm[row] = fminf(sqrtf(r1[0]), 1.0f);
}

// =====================================================================
// dt/tanh discretization: ad = 1 + 0.01*tanh(dt_raw@dt_w^T + dt_b)*(-tanh(a_log))
// bd = b_mat * xnorm ; cm = c_mat.  One thread per (row, ds).
// =====================================================================
__global__ __launch_bounds__(256) void k_dt(
    const float* __restrict__ p, const float* __restrict__ xnrm,
    const float* __restrict__ dtw, const float* __restrict__ dtb,
    const float* __restrict__ alog,
    float* __restrict__ ad, float* __restrict__ bd, float* __restrict__ cm)
{
    int idx = blockIdx.x * 256 + threadIdx.x;      // over Mrows*DSc
    int row = idx >> 4, ds = idx & 15;
    const float* pr = p + (size_t)row * (3 * DSc);
    float acc = dtb[ds];
    #pragma unroll
    for (int k = 0; k < DSc; ++k) acc += pr[2 * DSc + k] * dtw[ds * DSc + k];
    float dt = tanhf(acc) * 0.01f;
    float a = -tanhf(alog[ds]);
    ad[idx] = 1.0f + dt * a;
    bd[idx] = pr[ds] * xnrm[row];
    cm[idx] = pr[DSc + ds];
}

// =====================================================================
// Sequential SSM scan. One thread per (b, channel); DS=16 state in VGPRs.
// Per-step scalars (ad/bd/cm) staged to LDS in 64-step chunks.
// =====================================================================
__global__ __launch_bounds__(256) void k_scan(
    const float* __restrict__ xc,
    const float* __restrict__ ad, const float* __restrict__ bd,
    const float* __restrict__ cm, const float* __restrict__ dp,
    float* __restrict__ y)
{
    constexpr int CH = 64;                         // steps per LDS chunk
    __shared__ float sA[CH][DSc], sB[CH][DSc], sC[CH][DSc];
    const int tid = threadIdx.x;
    const int b = blockIdx.x >> 2;
    const int di = (blockIdx.x & 3) * 256 + tid;
    const float dpv = dp[di];

    float hst[DSc];
    #pragma unroll
    for (int k = 0; k < DSc; ++k) hst[k] = 0.f;

    for (int s0 = 0; s0 < Sc; s0 += CH) {
        __syncthreads();
        for (int i = tid; i < CH * DSc; i += 256) {
            int st = i >> 4, ds = i & 15;
            int g = (b * Sc + s0 + st) * DSc + ds;
            sA[st][ds] = ad[g]; sB[st][ds] = bd[g]; sC[st][ds] = cm[g];
        }
        __syncthreads();
        for (int sl = 0; sl < CH; ++sl) {
            int s = s0 + sl;
            float xt = xc[(size_t)(b * Sc + s) * DIc + di];
            float yacc = dpv * xt;
            #pragma unroll
            for (int ds = 0; ds < DSc; ++ds) {
                float hv = hst[ds] * sA[sl][ds] + sB[sl][ds] * xt;
                hv = fminf(fmaxf(hv, -10.0f), 10.0f);
                hst[ds] = hv;
                yacc += sC[sl][ds] * hv;
            }
            y[(size_t)(b * Sc + s) * DIc + di] = yacc;
        }
    }
}

// =====================================================================
// Gate: y *= silu(gate) with gate = proj[:, DI:2DI]
// =====================================================================
__global__ __launch_bounds__(256) void k_gate(
    const float* __restrict__ proj, float* __restrict__ y)
{
    int idx = blockIdx.x * 256 + threadIdx.x;      // over Mrows*DIc
    int row = idx / DIc, c = idx % DIc;
    float g = proj[(size_t)row * (2 * DIc) + DIc + c];
    y[idx] *= g / (1.0f + expf(-g));
}

// =====================================================================
// host-side orchestration
// =====================================================================
extern "C" void kernel_launch(void* const* d_in, const int* in_sizes, int n_in,
                              void* d_out, int out_size, void* d_ws, size_t ws_size,
                              hipStream_t stream)
{
    (void)in_sizes; (void)n_in; (void)out_size; (void)ws_size;

    const int*   x       = (const int*)  d_in[0];
    const int*   t       = (const int*)  d_in[1];
    const float* emb     = (const float*)d_in[2];
    const float* pos     = (const float*)d_in[3];
    const float* temb    = (const float*)d_in[4];
    const float* in_w    = (const float*)d_in[5];
    const float* conv_w  = (const float*)d_in[6];
    const float* conv_b  = (const float*)d_in[7];
    const float* ln_g    = (const float*)d_in[8];
    const float* ln_b    = (const float*)d_in[9];
    const float* xproj_w = (const float*)d_in[10];
    const float* dt_w    = (const float*)d_in[11];
    const float* dt_b    = (const float*)d_in[12];
    const float* a_log   = (const float*)d_in[13];
    const float* d_param = (const float*)d_in[14];
    const float* out_pw  = (const float*)d_in[15];
    const float* out_w   = (const float*)d_in[16];
    const float* out_b   = (const float*)d_in[17];
    float* out = (float*)d_out;
    float* ws  = (float*)d_ws;

    // workspace layout (floats): ~38.5 MB total
    float* h    = ws;                                  // 2048*512
    float* proj = h    + (size_t)Mrows * Dc;           // 2048*2048
    float* xc   = proj + (size_t)Mrows * 2 * DIc;      // 2048*1024
    float* y    = xc   + (size_t)Mrows * DIc;          // 2048*1024
    float* p    = y    + (size_t)Mrows * DIc;          // 2048*48
    float* xnrm = p    + (size_t)Mrows * 3 * DSc;      // 2048
    float* ad   = xnrm + Mrows;                        // 2048*16
    float* bd   = ad   + (size_t)Mrows * DSc;
    float* cm   = bd   + (size_t)Mrows * DSc;

    auto gemm = [&](const float* A, int lda, const float* Bm, int ldb,
                    float* C, int ldc, int Md, int Nd, int Kd,
                    const float* bias, const float* Cres) {
        dim3 grid((Nd + 63) / 64, (Md + 31) / 32);
        k_gemm_bf16x3<<<grid, dim3(256), 0, stream>>>(A, lda, Bm, ldb, C, ldc,
                                                      Md, Nd, Kd, bias, Cres);
    };

    k_embed<<<(Mrows * Dc) / 256, 256, 0, stream>>>(x, emb, pos, h);

    for (int l = 0; l < Lc; ++l) {
        // proj = h @ in_w[l]        (2048 x 512 x 2048)
        gemm(h, Dc, in_w + (size_t)l * Dc * 2 * DIc, 2 * DIc,
             proj, 2 * DIc, Mrows, 2 * DIc, Dc, nullptr, nullptr);
        // conv + temb + SiLU + LN + ||xc||
        k_conv_ln<<<Mrows, 256, 0, stream>>>(proj, t, temb,
                conv_w + (size_t)l * DIc * 4, conv_b + (size_t)l * DIc,
                ln_g + (size_t)l * DIc, ln_b + (size_t)l * DIc, xc, xnrm);
        // p = xc @ xproj_w[l]       (2048 x 1024 x 48)
        gemm(xc, DIc, xproj_w + (size_t)l * DIc * 3 * DSc, 3 * DSc,
             p, 3 * DSc, Mrows, 3 * DSc, DIc, nullptr, nullptr);
        // discretization
        k_dt<<<(Mrows * DSc) / 256, 256, 0, stream>>>(p, xnrm,
                dt_w + (size_t)l * DSc * DSc, dt_b + (size_t)l * DSc,
                a_log + (size_t)l * DSc, ad, bd, cm);
        // sequential scan
        k_scan<<<Bc * (DIc / 256), 256, 0, stream>>>(xc, ad, bd, cm,
                d_param + (size_t)l * DIc, y);
        // y *= silu(gate)
        k_gate<<<(Mrows * DIc) / 256, 256, 0, stream>>>(proj, y);
        // h += y @ out_proj_w[l]    (2048 x 1024 x 512), residual fused
        gemm(y, DIc, out_pw + (size_t)l * DIc * Dc, Dc,
             h, Dc, Mrows, Dc, DIc, nullptr, h);
    }

    // logits = h @ out_w + out_b    (2048 x 512 x 50257)
    gemm(h, Dc, out_w, Vc, out, Vc, Mrows, Vc, Dc, out_b, nullptr);
}
